// LSTMTagger_85212151152931
// MI455X (gfx1250) — matrix operand order repaired and checked by reading, not compiled, not run on hardware
//
#include <hip/hip_runtime.h>
#include <stdint.h>

// ---------------------------------------------------------------------------
// Problem constants (from reference)
// ---------------------------------------------------------------------------
#define S   8192
#define Wc  16
#define CD  32
#define Cc  128
#define E   256
#define Ii  384          // E + C
#define H   512
#define G4  2048         // 4*H
#define Tt  50

typedef __attribute__((ext_vector_type(16))) __bf16 v16bf;
typedef __attribute__((ext_vector_type(8)))  float  v8f;

// ---------------------------------------------------------------------------
// Helpers
// ---------------------------------------------------------------------------
__device__ __forceinline__ uint16_t f2bf(float f) {
    uint32_t u = __float_as_uint(f);
    uint32_t r = u + 0x7FFFu + ((u >> 16) & 1u);   // round-to-nearest-even
    return (uint16_t)(r >> 16);
}
__device__ __forceinline__ float bfl(uint32_t x) { return __uint_as_float(x << 16); }
__device__ __forceinline__ float bfh(uint32_t x) { return __uint_as_float(x & 0xFFFF0000u); }
__device__ __forceinline__ float sigf(float x)   { return 1.0f / (1.0f + __expf(-x)); }

struct FragBF16 { union { v16bf v; uint32_t u[8]; }; };

// ---------------------------------------------------------------------------
// 1) f32 -> bf16 weight conversion
// ---------------------------------------------------------------------------
__global__ void k_f32_to_bf16(const float* __restrict__ in, uint16_t* __restrict__ out, int n) {
    int i = blockIdx.x * 256 + threadIdx.x;
    if (i < n) out[i] = f2bf(in[i]);
}

// ---------------------------------------------------------------------------
// 2) Char CNN: per word (block) compute 128-channel max-pooled ReLU conv
// ---------------------------------------------------------------------------
__global__ void __launch_bounds__(128)
k_char_cnn(const int* __restrict__ chars, const float* __restrict__ cemb,
           const float* __restrict__ convw, const float* __restrict__ convb,
           float* __restrict__ feat) {
    int s = blockIdx.x;
    int c = threadIdx.x;                         // channel 0..127
    __shared__ __align__(16) float e[(Wc + 2) * CD];   // zero-padded char embeddings
    for (int i = threadIdx.x; i < (Wc + 2) * CD; i += 128) {
        int w = i / CD, d = i % CD;
        float v = 0.0f;
        if (w >= 1 && w <= Wc) {
            int ci = chars[s * Wc + (w - 1)];
            v = cemb[ci * CD + d];               // row CS is already all-zero
        }
        e[i] = v;
    }
    __syncthreads();
    const float* wc = convw + c * (3 * CD);
    float bias = convb[c];
    float best = 0.0f;                           // relu >= 0, so max >= 0
    for (int w = 0; w < Wc; ++w) {
        float acc = bias;
        const float* ew = &e[w * CD];            // window spans e[w..w+2][*], contiguous 96
        #pragma unroll 8
        for (int k = 0; k < 3 * CD; ++k) acc += ew[k] * wc[k];
        best = fmaxf(best, acc);
    }
    feat[s * Cc + c] = best;
}

// ---------------------------------------------------------------------------
// 3) Build x = [word_emb[sentence] | char_feat] in bf16
// ---------------------------------------------------------------------------
__global__ void k_build_x(const int* __restrict__ sentence, const float* __restrict__ wemb,
                          const float* __restrict__ feat, uint16_t* __restrict__ xbf) {
    size_t idx = (size_t)blockIdx.x * 256 + threadIdx.x;
    if (idx >= (size_t)S * Ii) return;
    int s = (int)(idx / Ii), i = (int)(idx % Ii);
    float v = (i < E) ? wemb[(size_t)sentence[s] * E + i] : feat[(size_t)s * Cc + (i - E)];
    xbf[idx] = f2bf(v);
}

// ---------------------------------------------------------------------------
// 4) Pre-activation GEMM (both directions via blockIdx.z):
//    pre[dir] = X(8192x384,bf16) @ Wih[dir].T + (bih+bhh), f32 accumulate.
//    Wave tile 16x64 (4x WMMA 16x16x32 bf16), block tile 32x256.
// ---------------------------------------------------------------------------
__global__ void __launch_bounds__(256)
k_pre_gemm(const uint16_t* __restrict__ X,
           const uint16_t* __restrict__ Wf, const uint16_t* __restrict__ Wb,
           const float* __restrict__ bihf, const float* __restrict__ bhhf,
           const float* __restrict__ bihb, const float* __restrict__ bhhb,
           float* __restrict__ pref, float* __restrict__ preb) {
    const uint16_t* Wt  = blockIdx.z ? Wb : Wf;    // (2048 x 384) row-major; B[k][n]=Wt[n][k]
    const float*    bi  = blockIdx.z ? bihb : bihf;
    const float*    bh  = blockIdx.z ? bhhb : bhhf;
    float*          out = blockIdx.z ? preb : pref;

    int wid  = threadIdx.x >> 5;                   // 0..7
    int lane = threadIdx.x & 31;
    int l16  = lane & 15, g = lane >> 4;

    int mBase = blockIdx.x * 32  + (wid & 1) * 16;
    int nBase = blockIdx.y * 256 + (wid >> 1) * 64;

    v8f acc[4] = {};
    for (int kb = 0; kb < Ii; kb += 32) {
        // ---- A fragment: lane l16 = row m; lane half g selects K sub-block ----
        FragBF16 a;
        const uint16_t* arow = X + (size_t)(mBase + l16) * Ii + kb + g * 8;
        #pragma unroll
        for (int v = 0; v < 8; ++v) {
            int k = (v < 4) ? (v * 2) : (8 + v * 2);   // K = {0..7,16..23} (+8 for g=1)
            a.u[v] = *(const uint32_t*)(arow + k);
        }
        // ---- 4 B fragments along N; lane l16 = column n, g selects K half ----
        #pragma unroll
        for (int t = 0; t < 4; ++t) {
            FragBF16 b;
            const uint16_t* brow = Wt + (size_t)(nBase + t * 16 + l16) * Ii + kb + g * 8;
            #pragma unroll
            for (int v = 0; v < 8; ++v) b.u[v] = *(const uint32_t*)(brow + ((v < 4) ? (v * 2) : (8 + v * 2)));
            acc[t] = __builtin_amdgcn_wmma_f32_16x16x32_bf16(
                false, a.v, false, b.v, (short)0, acc[t], false, false);
        }
    }
    // ---- Epilogue: D layout m = v + 8*g, n = l16; add fused bias ----
    #pragma unroll
    for (int t = 0; t < 4; ++t) {
        int n = nBase + t * 16 + l16;
        float bias = bi[n] + bh[n];
        #pragma unroll
        for (int v = 0; v < 8; ++v) {
            int m = mBase + v + g * 8;
            out[(size_t)m * G4 + n] = acc[t][v] + bias;
        }
    }
}

// ---------------------------------------------------------------------------
// 5) Recurrence: one persistent 1024-thread workgroup per direction.
//    h,c,gates live in LDS; Whh is bf16 (L2-resident). Thread owns gate rows
//    tid and tid+1024 (1024 FMAs/step), then lanes 0..511 do the cell update.
// ---------------------------------------------------------------------------
__global__ void __launch_bounds__(1024)
k_lstm_rec(const float* __restrict__ pre_f, const float* __restrict__ pre_b,
           const uint16_t* __restrict__ whh_f, const uint16_t* __restrict__ whh_b,
           float* __restrict__ hout) {
    int dir = blockIdx.x;                          // 0 = fwd, 1 = bwd
    const float*    pre = dir ? pre_b : pre_f;
    const uint16_t* whh = dir ? whh_b : whh_f;

    int tid = threadIdx.x;
    __shared__ __align__(16) float hs[H];
    __shared__ __align__(16) float cs[H];
    __shared__ __align__(16) float gts[G4];
    if (tid < H) { hs[tid] = 0.0f; cs[tid] = 0.0f; }
    __syncthreads();

    int r0 = tid, r1 = tid + 1024;
    const uint16_t* w0 = whh + (size_t)r0 * H;
    const uint16_t* w1 = whh + (size_t)r1 * H;

    for (int step = 0; step < S; ++step) {
        int t = dir ? (S - 1 - step) : step;
        const float* p = pre + (size_t)t * G4;
        float a0 = p[r0], a1 = p[r1];
        #pragma unroll 4
        for (int k = 0; k < H; k += 8) {
            float4 hv0 = *(const float4*)&hs[k];
            float4 hv1 = *(const float4*)&hs[k + 4];
            uint4  q0  = *(const uint4*)(w0 + k);
            uint4  q1  = *(const uint4*)(w1 + k);
            a0 += bfl(q0.x)*hv0.x + bfh(q0.x)*hv0.y + bfl(q0.y)*hv0.z + bfh(q0.y)*hv0.w
                + bfl(q0.z)*hv1.x + bfh(q0.z)*hv1.y + bfl(q0.w)*hv1.z + bfh(q0.w)*hv1.w;
            a1 += bfl(q1.x)*hv0.x + bfh(q1.x)*hv0.y + bfl(q1.y)*hv0.z + bfh(q1.y)*hv0.w
                + bfl(q1.z)*hv1.x + bfh(q1.z)*hv1.y + bfl(q1.w)*hv1.z + bfh(q1.w)*hv1.w;
        }
        gts[r0] = a0;
        gts[r1] = a1;
        __syncthreads();
        if (tid < H) {
            float ig = sigf(gts[tid]);
            float fg = sigf(gts[H + tid]);
            float gg = tanhf(gts[2 * H + tid]);
            float og = sigf(gts[3 * H + tid]);
            float cv = fg * cs[tid] + ig * gg;
            cs[tid] = cv;
            float hv = og * tanhf(cv);
            hs[tid] = hv;
            hout[(size_t)t * (2 * H) + dir * H + tid] = hv;
        }
        __syncthreads();
    }
}

// ---------------------------------------------------------------------------
// 6) Output projection: out = h_all(8192x1024) @ out_w.T + out_b
// ---------------------------------------------------------------------------
__global__ void __launch_bounds__(64)
k_out_proj(const float* __restrict__ hall, const float* __restrict__ ow,
           const float* __restrict__ ob, float* __restrict__ out) {
    int s  = blockIdx.x;
    int tg = threadIdx.x;                          // 0..63 (50 active)
    __shared__ __align__(16) float hrow[2 * H];
    for (int i = threadIdx.x; i < 2 * H; i += 64) hrow[i] = hall[(size_t)s * (2 * H) + i];
    __syncthreads();
    if (tg < Tt) {
        const float* w = ow + (size_t)tg * (2 * H);
        float acc = ob[tg];
        #pragma unroll 4
        for (int k = 0; k < 2 * H; k += 4) {
            float4 wv = *(const float4*)&w[k];
            float4 hv = *(const float4*)&hrow[k];
            acc += wv.x * hv.x + wv.y * hv.y + wv.z * hv.z + wv.w * hv.w;
        }
        out[(size_t)s * Tt + tg] = acc;
    }
}

// ---------------------------------------------------------------------------
// Launch
// ---------------------------------------------------------------------------
extern "C" void kernel_launch(void* const* d_in, const int* in_sizes, int n_in,
                              void* d_out, int out_size, void* d_ws, size_t ws_size,
                              hipStream_t stream) {
    (void)in_sizes; (void)n_in; (void)out_size; (void)ws_size;
    const int*   characters = (const int*)  d_in[0];
    const int*   sentence   = (const int*)  d_in[1];
    const float* char_emb   = (const float*)d_in[2];
    const float* conv_w     = (const float*)d_in[3];
    const float* conv_b     = (const float*)d_in[4];
    const float* word_emb   = (const float*)d_in[5];
    const float* Wih_f      = (const float*)d_in[6];
    const float* Whh_f      = (const float*)d_in[7];
    const float* bih_f      = (const float*)d_in[8];
    const float* bhh_f      = (const float*)d_in[9];
    const float* Wih_b      = (const float*)d_in[10];
    const float* Whh_b      = (const float*)d_in[11];
    const float* bih_b      = (const float*)d_in[12];
    const float* bhh_b      = (const float*)d_in[13];
    const float* out_w      = (const float*)d_in[14];
    const float* out_b      = (const float*)d_in[15];
    float*       out        = (float*)d_out;

    // Workspace carve-up (256B aligned)
    char* ws = (char*)d_ws;
    size_t off = 0;
    auto carve = [&](size_t bytes) -> char* {
        char* p = ws + off;
        off += (bytes + 255) & ~(size_t)255;
        return p;
    };
    uint16_t* xbf     = (uint16_t*)carve((size_t)S * Ii * 2);
    uint16_t* wihf_bf = (uint16_t*)carve((size_t)G4 * Ii * 2);
    uint16_t* wihb_bf = (uint16_t*)carve((size_t)G4 * Ii * 2);
    uint16_t* whhf_bf = (uint16_t*)carve((size_t)G4 * H * 2);
    uint16_t* whhb_bf = (uint16_t*)carve((size_t)G4 * H * 2);
    float*    feat    = (float*)   carve((size_t)S * Cc * 4);
    float*    pre_f   = (float*)   carve((size_t)S * G4 * 4);
    float*    pre_b   = (float*)   carve((size_t)S * G4 * 4);
    float*    hall    = (float*)   carve((size_t)S * 2 * H * 4);

    // 1) char CNN
    k_char_cnn<<<S, 128, 0, stream>>>(characters, char_emb, conv_w, conv_b, feat);

    // 2) weight conversions (f32 -> bf16)
    {
        int n1 = G4 * Ii, n2 = G4 * H;
        k_f32_to_bf16<<<(n1 + 255) / 256, 256, 0, stream>>>(Wih_f, wihf_bf, n1);
        k_f32_to_bf16<<<(n1 + 255) / 256, 256, 0, stream>>>(Wih_b, wihb_bf, n1);
        k_f32_to_bf16<<<(n2 + 255) / 256, 256, 0, stream>>>(Whh_f, whhf_bf, n2);
        k_f32_to_bf16<<<(n2 + 255) / 256, 256, 0, stream>>>(Whh_b, whhb_bf, n2);
    }

    // 3) build x (word emb gather + char feat concat, bf16)
    {
        size_t n = (size_t)S * Ii;
        k_build_x<<<(unsigned)((n + 255) / 256), 256, 0, stream>>>(sentence, word_emb, feat, xbf);
    }

    // 4) pre-activation GEMMs, both directions (WMMA bf16 -> f32)
    k_pre_gemm<<<dim3(S / 32, G4 / 256, 2), 256, 0, stream>>>(
        xbf, wihf_bf, wihb_bf, bih_f, bhh_f, bih_b, bhh_b, pre_f, pre_b);

    // 5) bidirectional recurrence (persistent workgroup per direction)
    k_lstm_rec<<<2, 1024, 0, stream>>>(pre_f, pre_b, whhf_bf, whhb_bf, hall);

    // 6) output projection
    k_out_proj<<<S, 64, 0, stream>>>(hall, out_w, out_b, out);
}
